// Attention_35304631173833
// MI455X (gfx1250) — matrix operand order repaired and checked by reading, not compiled, not run on hardware
//
#include <hip/hip_runtime.h>

#define DIM   768
#define NHEAD 12
#define HDIM  64
#define BATCH 4
#define SEQ   2048
#define MTOT  (BATCH * SEQ)   // 8192
#define SCALE 0.125f          // 64^-0.5

typedef __bf16 bf16_t;
typedef __attribute__((ext_vector_type(16))) bf16_t bf16x16;
typedef __attribute__((ext_vector_type(8)))  float  f32x8;
typedef __attribute__((ext_vector_type(4)))  unsigned int u32x4;

union FragU { u32x4 q[2]; bf16x16 f; };

__device__ __forceinline__ unsigned short f2bf(float f) {
  unsigned int u = __float_as_uint(f);
  u += 0x7FFFu + ((u >> 16) & 1u);          // round-to-nearest-even
  return (unsigned short)(u >> 16);
}

// A-fragment (16x32, M x K): lane = row (lane&15); elems 0..7 -> K = hi*8..hi*8+7,
// elems 8..15 -> K = hi*8+16..hi*8+23   (ISA 7.12.2, 16-bit A layout)
__device__ __forceinline__ bf16x16 load_a_frag(const unsigned short* tile, int ld) {
  const int lane = threadIdx.x & 31;
  const int row = lane & 15, hi = lane >> 4;
  const unsigned short* p = tile + row * ld + hi * 8;
  FragU u;
  u.q[0] = *(const u32x4*)(p);
  u.q[1] = *(const u32x4*)(p + 16);
  return u.f;
}

// B-fragment (32x16, K x N) from an N-major buffer (row n holds contraction-contiguous
// data): lane = column n (lane&15); elems 0..7 -> K = hi*8..hi*8+7, elems 8..15 -> K = hi*8+16..hi*8+23.
__device__ __forceinline__ bf16x16 load_b_frag(const unsigned short* tile, int ld) {
  const int lane = threadIdx.x & 31;
  const int n = lane & 15, hi = lane >> 4;
  const unsigned short* p = tile + n * ld + hi * 8;
  FragU u;
  u.q[0] = *(const u32x4*)(p);
  u.q[1] = *(const u32x4*)(p + 16);
  return u.f;
}

__device__ __forceinline__ f32x8 wmma_bf16(bf16x16 a, bf16x16 b, f32x8 c) {
  return __builtin_amdgcn_wmma_f32_16x16x32_bf16(false, a, false, b, (short)0, c,
                                                 false, false);
}

// ---------------------------------------------------------------- conversion
__global__ void cvt_f32_bf16(const float* __restrict__ src,
                             unsigned short* __restrict__ dst, int n) {
  int i = blockIdx.x * blockDim.x + threadIdx.x;
  if (i < n) dst[i] = f2bf(src[i]);
}

// ------------------------------------------------------------- QKV projection
// grid (12 heads, 128 row-tiles, 3 matrices), block 128 (4 waves x 16 rows)
__global__ __launch_bounds__(128) void qkv_kernel(
    const unsigned short* __restrict__ xb,
    const unsigned short* __restrict__ wq,
    const unsigned short* __restrict__ wk,
    const unsigned short* __restrict__ wv,
    unsigned short* __restrict__ qb,
    unsigned short* __restrict__ kb,
    unsigned short* __restrict__ vtb) {
  const int z = blockIdx.z;
  const unsigned short* w = (z == 0) ? wq : (z == 1) ? wk : wv;
  const int h  = blockIdx.x;                         // 64-wide col tile == head
  const int m0 = blockIdx.y * 64 + (threadIdx.x >> 5) * 16;
  const int lane = threadIdx.x & 31;
  const int n = lane & 15, hi = lane >> 4;

  f32x8 acc[4] = {};
  for (int kc = 0; kc < DIM; kc += 32) {
    bf16x16 a = load_a_frag(xb + (size_t)m0 * DIM + kc, DIM);
#pragma unroll
    for (int t = 0; t < 4; ++t) {
      bf16x16 b = load_b_frag(w + (size_t)(h * HDIM + t * 16) * DIM + kc, DIM);
      acc[t] = wmma_bf16(a, b, acc[t]);
    }
  }
#pragma unroll
  for (int t = 0; t < 4; ++t) {
#pragma unroll
    for (int j = 0; j < 8; ++j) {
      const int m = m0 + j + 8 * hi;
      const int b_ = m >> 11, nrow = m & (SEQ - 1);
      const int d = t * 16 + n;
      const unsigned short v = f2bf(acc[t][j]);
      if (z == 0)
        qb[(((size_t)(b_ * NHEAD + h) * SEQ) + nrow) * HDIM + d] = v;
      else if (z == 1)
        kb[(((size_t)(b_ * NHEAD + h) * SEQ) + nrow) * HDIM + d] = v;
      else
        vtb[(((size_t)(b_ * NHEAD + h) * HDIM) + d) * SEQ + nrow] = v;
    }
  }
}

// ------------------------------------------------------------ flash attention
// grid (32 q-tiles, 48 batch*head), block 128; wave w owns q rows [q0, q0+16)
__global__ __launch_bounds__(128) void attn_kernel(
    const unsigned short* __restrict__ qb,
    const unsigned short* __restrict__ kb,
    const unsigned short* __restrict__ vtb,
    const float* __restrict__ bias,
    unsigned short* __restrict__ ab) {
  __shared__ unsigned short Plds[4][16][32];
  const int bh = blockIdx.y;
  const int b_ = bh / NHEAD, h = bh % NHEAD;
  const int wave = threadIdx.x >> 5;
  const int q0 = blockIdx.x * 64 + wave * 16;
  const int lane = threadIdx.x & 31;
  const int n = lane & 15, hi = lane >> 4;

  const unsigned short* qh = qb  + (size_t)bh * SEQ * HDIM;
  const unsigned short* kh = kb  + (size_t)bh * SEQ * HDIM;
  const unsigned short* vh = vtb + (size_t)bh * HDIM * SEQ;
  const float* biash = bias + (size_t)h * SEQ * SEQ;

  const bf16x16 qa0 = load_a_frag(qh + (size_t)q0 * HDIM, HDIM);
  const bf16x16 qa1 = load_a_frag(qh + (size_t)q0 * HDIM + 32, HDIM);

  f32x8 o[4] = {};
  float rm[8], rl[8];
#pragma unroll
  for (int j = 0; j < 8; ++j) { rm[j] = -1e30f; rl[j] = 0.0f; }

  for (int kt = 0; kt < SEQ; kt += 32) {
    // Prefetch next key tile (global_prefetch_b8; no LOADcnt cost).
    // K tile: 32 rows x 128 B; lanes step 64 B -> only the first 16 rows (2 KB).
    if (kt + 32 < SEQ) {
      __builtin_prefetch(kh + (size_t)(kt + 32) * HDIM + (threadIdx.x & 31) * 32,
                         0, 3);
#pragma unroll
      for (int t = 0; t < 4; ++t)
        __builtin_prefetch(vh + (size_t)(t * 16 + n) * SEQ + (kt + 32) + hi * 16,
                           0, 3);
    }

    // S = Q @ K^T  (two 16-key sub-tiles, contraction d = 64 = 2 x 32)
    f32x8 s0 = {}, s1 = {};
    s0 = wmma_bf16(qa0, load_b_frag(kh + (size_t)kt * HDIM, HDIM), s0);
    s0 = wmma_bf16(qa1, load_b_frag(kh + (size_t)kt * HDIM + 32, HDIM), s0);
    s1 = wmma_bf16(qa0, load_b_frag(kh + (size_t)(kt + 16) * HDIM, HDIM), s1);
    s1 = wmma_bf16(qa1, load_b_frag(kh + (size_t)(kt + 16) * HDIM + 32, HDIM), s1);

    // scale + bias; bias is 201 MB read-once -> non-temporal so K/V stay in L2
#pragma unroll
    for (int j = 0; j < 8; ++j) {
      const float* brow = biash + (size_t)(q0 + j + 8 * hi) * SEQ + kt;
      s0[j] = s0[j] * SCALE + __builtin_nontemporal_load(brow + n);
      s1[j] = s1[j] * SCALE + __builtin_nontemporal_load(brow + 16 + n);
    }

    // online softmax (row stats live across the 16 lanes of each half)
    float al[8];
#pragma unroll
    for (int j = 0; j < 8; ++j) {
      float tm = fmaxf(s0[j], s1[j]);
      tm = fmaxf(tm, __shfl_xor(tm, 1, 32));
      tm = fmaxf(tm, __shfl_xor(tm, 2, 32));
      tm = fmaxf(tm, __shfl_xor(tm, 4, 32));
      tm = fmaxf(tm, __shfl_xor(tm, 8, 32));
      const float mnew = fmaxf(rm[j], tm);
      al[j] = __expf(rm[j] - mnew);
      rm[j] = mnew;
      s0[j] = __expf(s0[j] - mnew);
      s1[j] = __expf(s1[j] - mnew);
      float rs = s0[j] + s1[j];
      rs += __shfl_xor(rs, 1, 32);
      rs += __shfl_xor(rs, 2, 32);
      rs += __shfl_xor(rs, 4, 32);
      rs += __shfl_xor(rs, 8, 32);
      rl[j] = rl[j] * al[j] + rs;
    }
#pragma unroll
    for (int t = 0; t < 4; ++t)
#pragma unroll
      for (int j = 0; j < 8; ++j) o[t][j] *= al[j];

    // C-layout P -> A-layout P via per-wave LDS bounce (1 KB)
#pragma unroll
    for (int j = 0; j < 8; ++j) {
      Plds[wave][j + 8 * hi][n]      = f2bf(s0[j]);
      Plds[wave][j + 8 * hi][16 + n] = f2bf(s1[j]);
    }
    const bf16x16 pa = load_a_frag(&Plds[wave][0][0], 32);

    // O += P @ V   (V transposed => contiguous B-fragments)
#pragma unroll
    for (int t = 0; t < 4; ++t) {
      bf16x16 vb = load_b_frag(vh + (size_t)(t * 16) * SEQ + kt, SEQ);
      o[t] = wmma_bf16(pa, vb, o[t]);
    }
  }

#pragma unroll
  for (int j = 0; j < 8; ++j) rl[j] = 1.0f / rl[j];
#pragma unroll
  for (int t = 0; t < 4; ++t)
#pragma unroll
    for (int j = 0; j < 8; ++j) {
      const int nrow = q0 + j + 8 * hi;
      ab[((size_t)(b_ * SEQ + nrow)) * DIM + h * HDIM + t * 16 + n] =
          f2bf(o[t][j] * rl[j]);
    }
}

// ---------------------------------------------------------- output projection
// grid (12 col-tiles, 128 row-tiles), block 128
__global__ __launch_bounds__(128) void proj_kernel(
    const unsigned short* __restrict__ ab,
    const unsigned short* __restrict__ wp,
    const float* __restrict__ bpv,
    float* __restrict__ out) {
  const int c0 = blockIdx.x * 64;
  const int m0 = blockIdx.y * 64 + (threadIdx.x >> 5) * 16;
  const int lane = threadIdx.x & 31;
  const int n = lane & 15, hi = lane >> 4;

  f32x8 acc[4] = {};
  for (int kc = 0; kc < DIM; kc += 32) {
    bf16x16 a = load_a_frag(ab + (size_t)m0 * DIM + kc, DIM);
#pragma unroll
    for (int t = 0; t < 4; ++t) {
      bf16x16 b = load_b_frag(wp + (size_t)(c0 + t * 16) * DIM + kc, DIM);
      acc[t] = wmma_bf16(a, b, acc[t]);
    }
  }
  // Output written once, never re-read by this graph -> non-temporal stores.
#pragma unroll
  for (int t = 0; t < 4; ++t) {
    const float bias = bpv[c0 + t * 16 + n];
#pragma unroll
    for (int j = 0; j < 8; ++j) {
      const int m = m0 + j + 8 * hi;
      __builtin_nontemporal_store(acc[t][j] + bias,
                                  out + (size_t)m * DIM + c0 + t * 16 + n);
    }
  }
}

// ----------------------------------------------------------------- launcher
extern "C" void kernel_launch(void* const* d_in, const int* in_sizes, int n_in,
                              void* d_out, int out_size, void* d_ws, size_t ws_size,
                              hipStream_t stream) {
  const float* x    = (const float*)d_in[0];
  const float* bias = (const float*)d_in[1];
  const float* Wq   = (const float*)d_in[2];
  const float* Wk   = (const float*)d_in[3];
  const float* Wv   = (const float*)d_in[4];
  const float* Wp   = (const float*)d_in[5];
  const float* bp   = (const float*)d_in[6];
  float* out = (float*)d_out;

  char* ws = (char*)d_ws;
  size_t off = 0;
  auto alloc = [&](size_t bytes) -> void* {
    void* p = ws + off;
    off += (bytes + 255) & ~(size_t)255;
    return p;
  };
  unsigned short* xb  = (unsigned short*)alloc((size_t)MTOT * DIM * 2);
  unsigned short* wqb = (unsigned short*)alloc((size_t)DIM * DIM * 2);
  unsigned short* wkb = (unsigned short*)alloc((size_t)DIM * DIM * 2);
  unsigned short* wvb = (unsigned short*)alloc((size_t)DIM * DIM * 2);
  unsigned short* wpb = (unsigned short*)alloc((size_t)DIM * DIM * 2);
  unsigned short* qb  = (unsigned short*)alloc((size_t)MTOT * DIM * 2);
  unsigned short* kb  = (unsigned short*)alloc((size_t)MTOT * DIM * 2);
  unsigned short* vtb = (unsigned short*)alloc((size_t)MTOT * DIM * 2);
  unsigned short* ab  = (unsigned short*)alloc((size_t)MTOT * DIM * 2);

  const int nx = MTOT * DIM;
  const int nw = DIM * DIM;
  cvt_f32_bf16<<<(nx + 255) / 256, 256, 0, stream>>>(x, xb, nx);
  cvt_f32_bf16<<<(nw + 255) / 256, 256, 0, stream>>>(Wq, wqb, nw);
  cvt_f32_bf16<<<(nw + 255) / 256, 256, 0, stream>>>(Wk, wkb, nw);
  cvt_f32_bf16<<<(nw + 255) / 256, 256, 0, stream>>>(Wv, wvb, nw);
  cvt_f32_bf16<<<(nw + 255) / 256, 256, 0, stream>>>(Wp, wpb, nw);

  qkv_kernel<<<dim3(NHEAD, MTOT / 64, 3), 128, 0, stream>>>(xb, wqb, wkb, wvb,
                                                            qb, kb, vtb);
  attn_kernel<<<dim3(SEQ / 64, BATCH * NHEAD), 128, 0, stream>>>(qb, kb, vtb,
                                                                 bias, ab);
  proj_kernel<<<dim3(DIM / 64, MTOT / 64), 128, 0, stream>>>(ab, wpb, bp, out);
}
